// MLPPredictorEmbed_38087769981264
// MI455X (gfx1250) — matrix operand order repaired and checked by reading, not compiled, not run on hardware
//
#include <hip/hip_runtime.h>

typedef __attribute__((ext_vector_type(16))) _Float16 v16h;
typedef __attribute__((ext_vector_type(8)))  _Float16 v8h;
typedef __attribute__((ext_vector_type(4)))  _Float16 v4h;
typedef __attribute__((ext_vector_type(8)))  float    v8f;

#define EPB 128      // edges per block (M tile)
#define KD  256      // K = 2*IN_FEAT
#define ND  256      // N = EDGE_FEAT
#define AP  264      // padded A row length in halves  (528 B stride)
#define WP  264      // padded W1 row length in halves (528 B stride)
#define HP  260      // padded h_e row length in floats (1040 B stride)

// ---------------- prep: W1 f32 -> f16 (row-major [256][256]) ----------------
__global__ void __launch_bounds__(256)
cvt_w1_kernel(const float* __restrict__ W1, _Float16* __restrict__ W1h) {
  int i = (blockIdx.x * blockDim.x + threadIdx.x) * 4;   // 64 blocks * 256 thr * 4 = 65536
  float4 v = *(const float4*)(W1 + i);
  v4h o = {(_Float16)v.x, (_Float16)v.y, (_Float16)v.z, (_Float16)v.w};
  *(v4h*)(W1h + i) = o;
}

// ---------------- main fused kernel ----------------
__global__ void __launch_bounds__(256)
edge_mlp_kernel(const float* __restrict__ h,
                const int* __restrict__ src,
                const int* __restrict__ dst,
                const float* __restrict__ e,
                const _Float16* __restrict__ W1h,
                const float* __restrict__ W1b,
                const float* __restrict__ W2w,
                const float* __restrict__ W2b,
                float* __restrict__ out)
{
  // 67.6 KB A tile + 135.2 KB W1 tile = ~203 KB of the 320 KB WGP LDS.
  __shared__ __align__(16) _Float16 Alds[EPB * AP];
  __shared__ __align__(16) _Float16 Wlds[ND * WP];   // reused as f32 he[128][260] in phase 2/3

  const int tid = threadIdx.x;
  const int e0  = blockIdx.x * EPB;

  // ---- Phase 0a: async-DMA W1 (f16) into LDS, one 512B row per thread ----
  // GLOBAL_LOAD_ASYNC_TO_LDS_B128, GVS mode: lds[vdst] = mem[saddr + vaddr].
  // No VGPR data movement; tracked by ASYNCcnt. LDS byte address is the low
  // 32 bits of the generic pointer (flat-aperture mapping).
  {
    const unsigned lds0 = (unsigned)(size_t)(const void*)Wlds + (unsigned)(tid * (WP * 2));
    const unsigned gof0 = (unsigned)(tid * (KD * 2));
    #pragma unroll
    for (int i = 0; i < 32; ++i) {
      unsigned l = lds0 + i * 16;
      unsigned g = gof0 + i * 16;
      asm volatile("global_load_async_to_lds_b128 %0, %1, %2"
                   :: "v"(l), "v"(g), "s"(W1h) : "memory");
    }
  }

  // ---- Phase 0b: gather endpoint features, f32->f16, stage A = [h_u | h_v] ----
  {
    const int et   = tid >> 1;          // edge within block: 0..127
    const int half = tid & 1;           // 0: src row, 1: dst row
    const int node = half ? dst[e0 + et] : src[e0 + et];

    // Pull this block's e-rows toward the WGP while the GEMM runs (phase 3 uses them).
    __builtin_prefetch(e + (size_t)(e0 + et) * 32 + half * 16, 0, 0);

    const float4* hrow = (const float4*)(h + (long long)node * 128);
    _Float16* arow = Alds + et * AP + half * 128;
    #pragma unroll
    for (int i = 0; i < 16; ++i) {
      float4 a = hrow[2*i + 0];
      float4 b = hrow[2*i + 1];
      v8h c = {(_Float16)a.x,(_Float16)a.y,(_Float16)a.z,(_Float16)a.w,
               (_Float16)b.x,(_Float16)b.y,(_Float16)b.z,(_Float16)b.w};
      *(v8h*)(arow + 8*i) = c;
    }
  }

  // All async W1 transfers complete before anyone reads Wlds.
  asm volatile("s_wait_asynccnt 0x0" ::: "memory");
  __syncthreads();

  // ---- Phase 1: GEMM1  he[128][256] = A[128][256] x W1^T, f32 accumulate ----
  const int wave = tid >> 5, lane = tid & 31;
  const int wm  = (wave >> 2) * 64;    // wave M offset: 0 or 64
  const int wn  = (wave & 3) * 64;     // wave N offset: 0,64,128,192
  const int lhi = lane >> 4;           // lane half-select
  const int lm  = lane & 15;

  v8f acc[4][4] = {};                  // 4x4 grid of 16x16 f32 tiles = 128 VGPRs

  for (int k0 = 0; k0 < KD; k0 += 32) {
    // B fragments: lane holds column n = wn+tn*16+lm, 16 contiguous K at k0+lhi*16
    v16h bfrag[4];
    #pragma unroll
    for (int tn = 0; tn < 4; ++tn) {
      const _Float16* bp = Wlds + (wn + tn*16 + lm) * WP + k0 + lhi*16;
      v16h b;
      ((v8h*)&b)[0] = *(const v8h*)(bp);
      ((v8h*)&b)[1] = *(const v8h*)(bp + 8);
      bfrag[tn] = b;
    }
    #pragma unroll
    for (int tm = 0; tm < 4; ++tm) {
      // A fragment: lane holds row m = wm+tm*16+lm, 16 contiguous K at k0+lhi*16
      const _Float16* ap = Alds + (wm + tm*16 + lm) * AP + k0 + lhi*16;
      v16h a;
      ((v8h*)&a)[0] = *(const v8h*)(ap);
      ((v8h*)&a)[1] = *(const v8h*)(ap + 8);
      #pragma unroll
      for (int tn = 0; tn < 4; ++tn) {
        acc[tm][tn] = __builtin_amdgcn_wmma_f32_16x16x32_f16(
            /*neg_a=*/false, a, /*neg_b=*/false, bfrag[tn],
            /*c_mod=*/(short)0, acc[tm][tn], /*reuse_a=*/false, /*reuse_b=*/false);
      }
    }
  }

  __syncthreads();   // all waves finished reading Wlds -> safe to overwrite

  // ---- Phase 2: write h_e (+W1 bias) to LDS as f32, reusing the W1 region ----
  float* he = (float*)Wlds;            // [128][HP] f32 = 133.1 KB <= 135.2 KB
  #pragma unroll
  for (int tn = 0; tn < 4; ++tn) {
    const int n = wn + tn*16 + lm;
    const float bias = W1b[n];
    #pragma unroll
    for (int tm = 0; tm < 4; ++tm) {
      const int mb = wm + tm*16 + lhi*8;   // C layout: VGPR r -> row mb+r, col n
      v8f c = acc[tm][tn];
      #pragma unroll
      for (int r = 0; r < 8; ++r)
        he[(mb + r) * HP + n] = c[r] + bias;
    }
  }
  __syncthreads();

  // ---- Phase 3: GEMM2 epilogue, one thread per (edge, class) ----
  {
    const int et  = tid >> 1;
    const int cls = tid & 1;
    const float4* herow = (const float4*)(he + et * HP);
    const float4* w2    = (const float4*)(W2w + cls * 288);
    float4 sv = {0.f, 0.f, 0.f, 0.f};
    #pragma unroll
    for (int i = 0; i < 64; ++i) {       // dot(h_e_row, W2a_row)
      float4 x = herow[i], w = w2[i];
      sv.x += x.x*w.x; sv.y += x.y*w.y; sv.z += x.z*w.z; sv.w += x.w*w.w;
    }
    const float4* erow = (const float4*)(e + (long long)(e0 + et) * 32);
    #pragma unroll
    for (int i = 0; i < 8; ++i) {        // dot(e_row, W2b_row)
      float4 x = erow[i], w = w2[64 + i];
      sv.x += x.x*w.x; sv.y += x.y*w.y; sv.z += x.z*w.z; sv.w += x.w*w.w;
    }
    float s = W2b[cls] + sv.x + sv.y + sv.z + sv.w;
    out[(long long)(e0 + et) * 2 + cls] = s;
  }
}

extern "C" void kernel_launch(void* const* d_in, const int* in_sizes, int n_in,
                              void* d_out, int out_size, void* d_ws, size_t ws_size,
                              hipStream_t stream) {
  const float* h    = (const float*)d_in[0];
  const int*   src  = (const int*)d_in[1];
  const int*   dst  = (const int*)d_in[2];
  const float* e    = (const float*)d_in[3];
  const float* W1w  = (const float*)d_in[4];
  const float* W1b  = (const float*)d_in[5];
  const float* W2w  = (const float*)d_in[6];
  const float* W2b  = (const float*)d_in[7];
  float* out = (float*)d_out;

  _Float16* W1h = (_Float16*)d_ws;       // 128 KB scratch: f16 copy of W1

  cvt_w1_kernel<<<64, 256, 0, stream>>>(W1w, W1h);

  const int nEdges = in_sizes[1];        // 640000
  edge_mlp_kernel<<<nEdges / EPB, 256, 0, stream>>>(
      h, src, dst, e, W1h, W1b, W2w, W2b, out);
}
